// MultiHeadLatentAttention_23450521436545
// MI455X (gfx1250) — compile-verified, matrix operand order checked by reading, not run on hardware
//
#include <hip/hip_runtime.h>
#include <hip/hip_bf16.h>

// ---------------------------------------------------------------------------
// MLA forward for MI455X (gfx1250, wave32, WMMA, async-to-LDS staging).
// Pipeline (bf16 operands everywhere, f32 WMMA accumulation):
//   0. xb  = bf16(x);  Wt_* = bf16(W_*^T)      (one-time conversion kernels)
//   1. latents = xb @ W_down + b_down          (GEMM, bf16 out)
//   2. keys    = latents @ W_uk + b_uk         (GEMM, bf16 out)
//   3. values  = latents @ W_uv + b_uv         (GEMM, bf16 out)
//   4. q       = xb @ W_q + b_q                (GEMM, bf16 out)
//   5. ctx     = softmax(q k^T / sqrt(128)) v  (flash attention, WMMA)
//   6. out     = ctx @ W_o + b_o               (GEMM, f32 out)
// Workspace requirement: ~200 MB.
// ---------------------------------------------------------------------------

#ifndef USE_ASYNC_LDS
#define USE_ASYNC_LDS 1   // flip to 0 to fall back to synchronous staging
#endif

typedef __attribute__((ext_vector_type(16))) __bf16 v16bf;
typedef __attribute__((ext_vector_type(8)))  float  v8f;

union Frag16 { v16bf v; uint4 q[2]; };

static constexpr int MTOT   = 8192;   // B*S
static constexpr int DMODEL = 2048;
static constexpr int DLAT   = 512;
static constexpr int NH     = 16;
static constexpr int DH     = 128;
static constexpr int SEQ    = 2048;

__device__ __forceinline__ unsigned short f2bf(float f) {
  unsigned int u = __builtin_bit_cast(unsigned int, f);
  u = (u + 0x7FFFu + ((u >> 16) & 1u)) >> 16;   // round-to-nearest-even
  return (unsigned short)u;
}

// 16-byte async copy global -> LDS (ISA 10. GLOBAL_LOAD_ASYNC_TO_LDS_B128,
// tracked by ASYNCcnt). For a generic pointer to __shared__, addr[31:0] is the
// wave-relative LDS byte offset (ISA 10.2 aperture mapping), which is exactly
// what the VDST VGPR of the async op must hold.
__device__ __forceinline__ void async_b128(void* ldsp, const void* g) {
#if USE_ASYNC_LDS
  const unsigned int loff = (unsigned int)(uintptr_t)ldsp;
  asm volatile("global_load_async_to_lds_b128 %0, %1, off"
               :: "v"(loff), "v"(g) : "memory");
#else
  *(uint4*)ldsp = *(const uint4*)g;
#endif
}

__device__ __forceinline__ void wait_async0() {
#if USE_ASYNC_LDS
#if defined(__has_builtin) && __has_builtin(__builtin_amdgcn_s_wait_asynccnt)
  __builtin_amdgcn_s_wait_asynccnt(0);
#else
  asm volatile("s_wait_asynccnt 0x0" ::: "memory");
#endif
#endif
}

// Load one 16x32 bf16 WMMA fragment row for this lane.
// Per ISA 7.12.2 (16-bit A-matrix 16x32): lane group g = lane/16 holds
// K = 8g..8g+7 (VGPR0-3) and K = 16+8g..16+8g+7 (VGPR4-7), packed lo/hi.
// Caller passes p = rowBase + 8*(lane>>4); we read p[0..7] and p[16..23].
__device__ __forceinline__ v16bf load_frag(const unsigned short* p) {
  Frag16 f;
  f.q[0] = *(const uint4*)(p);
  f.q[1] = *(const uint4*)(p + 16);
  return f.v;
}

// ---------------------------------------------------------------------------
// One-time operand preparation kernels
// ---------------------------------------------------------------------------
__global__ __launch_bounds__(256)
void conv_f32_bf16(const float* __restrict__ src, unsigned short* __restrict__ dst) {
  const size_t i = ((size_t)blockIdx.x * 256 + threadIdx.x) * 8;
  const float4 f0 = *(const float4*)(src + i);
  const float4 f1 = *(const float4*)(src + i + 4);
  unsigned short t[8];
  t[0] = f2bf(f0.x); t[1] = f2bf(f0.y); t[2] = f2bf(f0.z); t[3] = f2bf(f0.w);
  t[4] = f2bf(f1.x); t[5] = f2bf(f1.y); t[6] = f2bf(f1.z); t[7] = f2bf(f1.w);
  *(uint4*)(dst + i) = *(uint4*)t;
}

// W [K,N] f32 -> Wt [N,K] bf16, 32x32 LDS-tiled transpose, 256 threads (32x8).
__global__ __launch_bounds__(256)
void transpose_w_bf16(const float* __restrict__ W, unsigned short* __restrict__ Wt,
                      int K, int N) {
  __shared__ float t[32][33];
  const int n0 = blockIdx.x * 32, k0 = blockIdx.y * 32;
  const int tx = threadIdx.x & 31, ty = threadIdx.x >> 5;
#pragma unroll
  for (int j = 0; j < 32; j += 8)
    t[ty + j][tx] = W[(size_t)(k0 + ty + j) * N + n0 + tx];
  __syncthreads();
#pragma unroll
  for (int j = 0; j < 32; j += 8)
    Wt[(size_t)(n0 + ty + j) * K + k0 + tx] = f2bf(t[tx][ty + j]);
}

// ---------------------------------------------------------------------------
// Tiled GEMM: C[M,N] = A[M,K] @ Bt[N,K]^T + bias[N]   (A, Bt bf16)
// Block: 256 threads (8 waves), tile 128(M) x 128(N), K step 32.
// Wave grid 4x2, each wave owns a 32x64 tile = 2x4 WMMA 16x16 tiles.
// Double-buffered LDS staged with async-to-LDS copies.
// ---------------------------------------------------------------------------
template<bool OUT_BF16>
__global__ __launch_bounds__(256)
void gemm_bias_wmma(const unsigned short* __restrict__ A,
                    const unsigned short* __restrict__ Bt,
                    const float* __restrict__ bias, void* __restrict__ Cv,
                    int M, int N, int K) {
  __shared__ unsigned short As[2][128 * 32];   // [row][k] bf16
  __shared__ unsigned short Bs[2][128 * 32];   // [col][k] bf16

  const int tid  = threadIdx.x;
  const int lane = tid & 31;
  const int wv   = tid >> 5;
  const int wm   = wv >> 1;             // 0..3 -> 32-row slice
  const int wn   = wv & 1;              // 0..1 -> 64-col slice
  const int m0   = blockIdx.y * 128;
  const int n0   = blockIdx.x * 128;
  const int l16  = lane & 15;
  const int kb   = (lane >> 4) * 8;
  const int srow = tid >> 1;            // 0..127 staging row
  const int soff = (tid & 1) * 16;      // 0 or 16 (elements)

  v8f acc[2][4] = {};
  const int ksteps = K >> 5;

  auto stage = [&](int kt, int buf) {
    const unsigned short* ga = A + (size_t)(m0 + srow) * K + kt * 32 + soff;
    async_b128(&As[buf][srow * 32 + soff],     ga);
    async_b128(&As[buf][srow * 32 + soff + 8], ga + 8);
    const unsigned short* gb = Bt + (size_t)(n0 + srow) * K + kt * 32 + soff;
    async_b128(&Bs[buf][srow * 32 + soff],     gb);
    async_b128(&Bs[buf][srow * 32 + soff + 8], gb + 8);
  };

  stage(0, 0);
  for (int kt = 0; kt < ksteps; ++kt) {
    const int cur = kt & 1;
    wait_async0();        // own async copies for tile kt complete
    __syncthreads();      // all waves' copies visible; prior reads done
    if (kt + 1 < ksteps) stage(kt + 1, cur ^ 1);

    v16bf af[2], bfr[4];
#pragma unroll
    for (int mi = 0; mi < 2; ++mi)
      af[mi] = load_frag(&As[cur][(wm * 32 + mi * 16 + l16) * 32 + kb]);
#pragma unroll
    for (int ni = 0; ni < 4; ++ni)
      bfr[ni] = load_frag(&Bs[cur][(wn * 64 + ni * 16 + l16) * 32 + kb]);

#pragma unroll
    for (int mi = 0; mi < 2; ++mi)
#pragma unroll
      for (int ni = 0; ni < 4; ++ni)
        acc[mi][ni] = __builtin_amdgcn_wmma_f32_16x16x32_bf16(
            false, af[mi], false, bfr[ni], (short)0, acc[mi][ni], false, false);
  }

  // epilogue: C-layout lane = n%16, vgpr r -> row r + 8*(lane/16)
#pragma unroll
  for (int mi = 0; mi < 2; ++mi) {
#pragma unroll
    for (int ni = 0; ni < 4; ++ni) {
      const int col  = n0 + wn * 64 + ni * 16 + l16;
      const float bv = bias[col];
#pragma unroll
      for (int r = 0; r < 8; ++r) {
        const int row   = m0 + wm * 32 + mi * 16 + r + 8 * (lane >> 4);
        const float val = acc[mi][ni][r] + bv;
        if (OUT_BF16)
          ((unsigned short*)Cv)[(size_t)row * N + col] = f2bf(val);
        else
          ((float*)Cv)[(size_t)row * N + col] = val;
      }
    }
  }
}

// ---------------------------------------------------------------------------
// Flash attention. Block = 256 threads (8 waves), wave = 16 query rows,
// block covers 128 query rows of one (b,h). 32-key tiles, online softmax.
// ---------------------------------------------------------------------------
__global__ __launch_bounds__(256)
void mla_attn_wmma(const unsigned short* __restrict__ Q,
                   const unsigned short* __restrict__ Kb,
                   const unsigned short* __restrict__ Vb,
                   unsigned short* __restrict__ Ctx) {
  __shared__ unsigned short Kls[32 * 128];      // [key][d]
  __shared__ unsigned short Vt[128 * 32];       // [d][key]  (transposed)
  __shared__ unsigned short Pst[8 * 16 * 32];   // per-wave P staging [m][k]

  const int tid  = threadIdx.x;
  const int lane = tid & 31;
  const int wv   = tid >> 5;
  const int b    = blockIdx.y >> 4;
  const int h    = blockIdx.y & 15;
  const int qbase = blockIdx.x * 128 + wv * 16;
  const int l16  = lane & 15;
  const int kb   = (lane >> 4) * 8;
  const size_t rowstride = (size_t)NH * DH;     // 2048
  const size_t headoff   = (size_t)h * DH;
  const size_t batchbase = (size_t)b * SEQ;
  const float  scale     = 0.08838834764831845f; // 1/sqrt(128)

  // Q fragments: 16 rows x 128 dims = 4 WMMA K-steps, loaded once from global.
  v16bf qf[4];
  {
    const unsigned short* qrow =
        Q + (batchbase + qbase + l16) * rowstride + headoff;
#pragma unroll
    for (int d = 0; d < 4; ++d)
      qf[d] = load_frag(qrow + d * 32 + kb);
  }

  float rmax[8], rsum[8];
#pragma unroll
  for (int r = 0; r < 8; ++r) { rmax[r] = -3.0e38f; rsum[r] = 0.0f; }
  v8f acc[8] = {};
  unsigned short* stage = &Pst[wv * 16 * 32];

  const int lkey = tid >> 3;           // 0..31 key within tile
  const int ldc  = (tid & 7) * 16;     // 0..112 dim chunk

  for (int kt = 0; kt < SEQ / 32; ++kt) {
    __syncthreads();   // protect last iter's LDS reads before overwrite
    {
      const size_t srow =
          (batchbase + (size_t)kt * 32 + lkey) * rowstride + headoff + ldc;
      // K tile row-major: pure copy -> async to LDS
      const unsigned short* ksrc = Kb + srow;
      async_b128(&Kls[lkey * 128 + ldc],     ksrc);
      async_b128(&Kls[lkey * 128 + ldc + 8], ksrc + 8);
      // V tile transposed (element scatter, manual)
      unsigned short tmp[16];
      *(uint4*)&tmp[0] = ((const uint4*)(Vb + srow))[0];
      *(uint4*)&tmp[8] = ((const uint4*)(Vb + srow))[1];
#pragma unroll
      for (int j = 0; j < 16; ++j)
        Vt[(ldc + j) * 32 + lkey] = tmp[j];
    }
    wait_async0();
    __syncthreads();

    // scores: S = Q(16x128) @ K_tile^T(128x32) -> two 16x16 f32 tiles
    v8f s0 = {}, s1 = {};
#pragma unroll
    for (int d = 0; d < 4; ++d) {
      v16bf b0 = load_frag(&Kls[l16 * 128 + d * 32 + kb]);
      v16bf b1 = load_frag(&Kls[(16 + l16) * 128 + d * 32 + kb]);
      s0 = __builtin_amdgcn_wmma_f32_16x16x32_bf16(false, qf[d], false, b0,
                                                   (short)0, s0, false, false);
      s1 = __builtin_amdgcn_wmma_f32_16x16x32_bf16(false, qf[d], false, b1,
                                                   (short)0, s1, false, false);
    }

    // online softmax: row m = r + 8*(lane/16), 16 lanes share a row
    float corr[8];
#pragma unroll
    for (int r = 0; r < 8; ++r) {
      float a0 = s0[r] * scale, a1 = s1[r] * scale;
      float mx = fmaxf(a0, a1);
      mx = fmaxf(mx, __shfl_xor(mx, 1, 32));
      mx = fmaxf(mx, __shfl_xor(mx, 2, 32));
      mx = fmaxf(mx, __shfl_xor(mx, 4, 32));
      mx = fmaxf(mx, __shfl_xor(mx, 8, 32));
      const float nm = fmaxf(rmax[r], mx);
      const float c  = __expf(rmax[r] - nm);
      const float p0 = __expf(a0 - nm);
      const float p1 = __expf(a1 - nm);
      float ps = p0 + p1;
      ps += __shfl_xor(ps, 1, 32);
      ps += __shfl_xor(ps, 2, 32);
      ps += __shfl_xor(ps, 4, 32);
      ps += __shfl_xor(ps, 8, 32);
      rsum[r] = rsum[r] * c + ps;
      rmax[r] = nm;
      corr[r] = c;
      const int prow = r + 8 * (lane >> 4);
      stage[prow * 32 + l16]      = f2bf(p0);   // keys 0..15
      stage[prow * 32 + 16 + l16] = f2bf(p1);   // keys 16..31
    }
#pragma unroll
    for (int nc = 0; nc < 8; ++nc)
#pragma unroll
      for (int r = 0; r < 8; ++r)
        acc[nc][r] *= corr[r];
    __syncthreads();

    // ctx += P(16x32) @ V_tile(32x128): 8 WMMAs over dim chunks
    v16bf pf = load_frag(&stage[l16 * 32 + kb]);
#pragma unroll
    for (int nc = 0; nc < 8; ++nc) {
      v16bf vf = load_frag(&Vt[(nc * 16 + l16) * 32 + kb]);
      acc[nc] = __builtin_amdgcn_wmma_f32_16x16x32_bf16(
          false, pf, false, vf, (short)0, acc[nc], false, false);
    }
  }

  // normalize + store ctx as bf16 [B*S, H*D]
  float inv[8];
#pragma unroll
  for (int r = 0; r < 8; ++r) inv[r] = 1.0f / rsum[r];
#pragma unroll
  for (int nc = 0; nc < 8; ++nc) {
#pragma unroll
    for (int r = 0; r < 8; ++r) {
      const int row    = qbase + r + 8 * (lane >> 4);
      const size_t col = headoff + nc * 16 + l16;
      Ctx[(batchbase + row) * rowstride + col] = f2bf(acc[nc][r] * inv[r]);
    }
  }
}

// ---------------------------------------------------------------------------
extern "C" void kernel_launch(void* const* d_in, const int* in_sizes, int n_in,
                              void* d_out, int out_size, void* d_ws, size_t ws_size,
                              hipStream_t stream) {
  const float* x      = (const float*)d_in[0];
  const float* W_down = (const float*)d_in[1];
  const float* b_down = (const float*)d_in[2];
  const float* W_uk   = (const float*)d_in[3];
  const float* b_uk   = (const float*)d_in[4];
  const float* W_uv   = (const float*)d_in[5];
  const float* b_uv   = (const float*)d_in[6];
  const float* W_q    = (const float*)d_in[7];
  const float* b_q    = (const float*)d_in[8];
  const float* W_o    = (const float*)d_in[9];
  const float* b_o    = (const float*)d_in[10];
  float* out = (float*)d_out;

  char* ws = (char*)d_ws;
  auto take = [&](size_t elems) {
    unsigned short* p = (unsigned short*)ws;
    ws += elems * 2;
    return p;
  };
  unsigned short* xb  = take((size_t)MTOT * DMODEL);
  unsigned short* WtD = take((size_t)DMODEL * DLAT);    // [DLAT, DMODEL]
  unsigned short* WtK = take((size_t)DLAT * DMODEL);    // [DMODEL, DLAT]
  unsigned short* WtV = take((size_t)DLAT * DMODEL);
  unsigned short* WtQ = take((size_t)DMODEL * DMODEL);
  unsigned short* WtO = take((size_t)DMODEL * DMODEL);
  unsigned short* lat = take((size_t)MTOT * DLAT);
  unsigned short* qb  = take((size_t)MTOT * DMODEL);
  unsigned short* kbf = take((size_t)MTOT * DMODEL);
  unsigned short* vbf = take((size_t)MTOT * DMODEL);
  unsigned short* ctx = take((size_t)MTOT * DMODEL);

  const dim3 blk(256);

  // 0. operand preparation
  conv_f32_bf16<<<(MTOT * (size_t)DMODEL) / (256 * 8), blk, 0, stream>>>(x, xb);
  transpose_w_bf16<<<dim3(DLAT / 32, DMODEL / 32), blk, 0, stream>>>(W_down, WtD, DMODEL, DLAT);
  transpose_w_bf16<<<dim3(DMODEL / 32, DLAT / 32), blk, 0, stream>>>(W_uk, WtK, DLAT, DMODEL);
  transpose_w_bf16<<<dim3(DMODEL / 32, DLAT / 32), blk, 0, stream>>>(W_uv, WtV, DLAT, DMODEL);
  transpose_w_bf16<<<dim3(DMODEL / 32, DMODEL / 32), blk, 0, stream>>>(W_q, WtQ, DMODEL, DMODEL);
  transpose_w_bf16<<<dim3(DMODEL / 32, DMODEL / 32), blk, 0, stream>>>(W_o, WtO, DMODEL, DMODEL);

  const dim3 gLat(DLAT / 128,   MTOT / 128);
  const dim3 gDM (DMODEL / 128, MTOT / 128);

  // 1. latents = xb @ W_down + b_down
  gemm_bias_wmma<true><<<gLat, blk, 0, stream>>>(xb, WtD, b_down, lat, MTOT, DLAT, DMODEL);
  // 2. keys = latents @ W_uk + b_uk
  gemm_bias_wmma<true><<<gDM, blk, 0, stream>>>(lat, WtK, b_uk, kbf, MTOT, DMODEL, DLAT);
  // 3. values = latents @ W_uv + b_uv
  gemm_bias_wmma<true><<<gDM, blk, 0, stream>>>(lat, WtV, b_uv, vbf, MTOT, DMODEL, DLAT);
  // 4. q = xb @ W_q + b_q
  gemm_bias_wmma<true><<<gDM, blk, 0, stream>>>(xb, WtQ, b_q, qb, MTOT, DMODEL, DMODEL);
  // 5. attention
  mla_attn_wmma<<<dim3(SEQ / 128, 4 * NH), blk, 0, stream>>>(qb, kbf, vbf, ctx);
  // 6. out = ctx @ W_o + b_o
  gemm_bias_wmma<false><<<gDM, blk, 0, stream>>>(ctx, WtO, b_o, out, MTOT, DMODEL, DMODEL);
}
